// TR3MotifNet_75720273429180
// MI455X (gfx1250) — compile-verified, matrix operand order checked by reading, not run on hardware
//
#include <hip/hip_runtime.h>
#include <hip/hip_bf16.h>

typedef _Float16 v16h __attribute__((ext_vector_type(16)));
typedef _Float16 v8h  __attribute__((ext_vector_type(8)));
typedef float    v8f  __attribute__((ext_vector_type(8)));

#define NN 100000
#define EE 1600000
#define GG 512
#define DD 64

// Hardware f32 atomic add (global_atomic_add_f32), avoiding any CAS-loop
// lowering of plain atomicAdd(float*).
__device__ __forceinline__ void atomAddF32(float* p, float v) {
  unsafeAtomicAdd(p, v);
}

// -------------------- embed: h = x @ W_emb + b_emb (also f16 copy) ----------
__global__ __launch_bounds__(256) void embed_kernel(
    const float* __restrict__ x, const float* __restrict__ Wemb,
    const float* __restrict__ bemb, float* __restrict__ h,
    _Float16* __restrict__ h16, int N) {
  int idx = blockIdx.x * blockDim.x + threadIdx.x;
  if (idx >= N * DD) return;
  int n = idx >> 6, d = idx & 63;
  float v = bemb[d];
#pragma unroll
  for (int k = 0; k < 4; ++k) v += x[n * 4 + k] * Wemb[k * DD + d];
  h[idx] = v;
  h16[idx] = (_Float16)v;
}

// -------------------- wdeg[i] = sum of edge_attr over incoming edges --------
__global__ __launch_bounds__(256) void wdeg_kernel(
    const int* __restrict__ ei, const float* __restrict__ ea,
    float* __restrict__ wdeg, int E) {
  int e = blockIdx.x * blockDim.x + threadIdx.x;
  if (e >= E) return;
  atomAddF32(&wdeg[ei[E + e]], ea[e]);
}

// ---- pack all layer weights (W1,W2,W3) into WMMA B-fragment order (f16) ----
// wpack[(((layer*3+mat)*2+kc)*4+nt)*512 + lane*16 + e]
__global__ __launch_bounds__(256) void pack_weights_kernel(
    const float* __restrict__ W1, const float* __restrict__ W2,
    const float* __restrict__ W3, _Float16* __restrict__ wpack) {
  int t = blockIdx.x * blockDim.x + threadIdx.x;
  const int total = 3 * 3 * 2 * 4 * 32;  // layer*mat*kc*nt*lane
  if (t >= total) return;
  int layer = t / 768;
  int r = t % 768;
  int mat = r / 256;
  int r2 = r % 256;
  int kc = r2 / 128;
  int r3 = r2 % 128;
  int nt = r3 / 32;
  int lane = r3 % 32;
  const float* W = (mat == 0 ? W1 : (mat == 1 ? W2 : W3)) + layer * DD * DD;
  int n = nt * 16 + (lane & 15);
  int khi = lane >> 4;
  _Float16* out = wpack + ((((size_t)(layer * 3 + mat) * 2 + kc) * 4 + nt) * 512)
                        + lane * 16;
#pragma unroll
  for (int e = 0; e < 16; ++e) {
    int k = kc * 32 + (e >> 3) * 16 + khi * 8 + (e & 7);
    out[e] = (_Float16)W[k * DD + n];
  }
}

// ------- one wave per 16-row tile: a = h@W1+b1; agg = h@W3+b3 - wdeg*(h@W2) -
__global__ __launch_bounds__(256) void gemm3_kernel(
    const _Float16* __restrict__ h16, const _Float16* __restrict__ wpack,
    const float* __restrict__ b1, const float* __restrict__ b3,
    const float* __restrict__ wdeg, float* __restrict__ a_out,
    float* __restrict__ agg_out, int layer, int ntiles) {
  int wave = (blockIdx.x * blockDim.x + threadIdx.x) >> 5;
  int lane = threadIdx.x & 31;
  if (wave >= ntiles) return;  // wave-uniform: EXEC stays all-1s for WMMA
  int row0 = wave * 16;
  int mlo = lane & 15;
  int khi = lane >> 4;

  // A fragments: 16x32 f16, two K-chunks covering K=64
  v16h afrag[2];
#pragma unroll
  for (int kc = 0; kc < 2; ++kc) {
    const _Float16* ap = h16 + (size_t)(row0 + mlo) * DD + kc * 32 + khi * 8;
    v8h lo = *(const v8h*)ap;         // K = kbase .. kbase+7
    v8h hi = *(const v8h*)(ap + 16);  // K = 16+kbase .. 16+kbase+7
    v16h f;
#pragma unroll
    for (int i = 0; i < 8; ++i) { f[i] = lo[i]; f[i + 8] = hi[i]; }
    afrag[kc] = f;
  }

  const _Float16* wl = wpack + (size_t)layer * 12288;
  float wd[8];
#pragma unroll
  for (int r = 0; r < 8; ++r) wd[r] = wdeg[row0 + khi * 8 + r];

#pragma unroll
  for (int nt = 0; nt < 4; ++nt) {
    int n = nt * 16 + mlo;
    v8f acc1 = {}, acc2 = {}, acc3 = {};
#pragma unroll
    for (int kc = 0; kc < 2; ++kc) {
      v16h bf1 = *(const v16h*)(wl + (((0 * 2 + kc) * 4 + nt) * 512) + lane * 16);
      acc1 = __builtin_amdgcn_wmma_f32_16x16x32_f16(false, afrag[kc], false, bf1,
                                                    (short)0, acc1, false, false);
      v16h bf2 = *(const v16h*)(wl + (((1 * 2 + kc) * 4 + nt) * 512) + lane * 16);
      acc2 = __builtin_amdgcn_wmma_f32_16x16x32_f16(false, afrag[kc], false, bf2,
                                                    (short)0, acc2, false, false);
      v16h bf3 = *(const v16h*)(wl + (((2 * 2 + kc) * 4 + nt) * 512) + lane * 16);
      acc3 = __builtin_amdgcn_wmma_f32_16x16x32_f16(false, afrag[kc], false, bf3,
                                                    (short)0, acc3, false, false);
    }
    float bias1 = b1[layer * DD + n];
    float bias3 = b3[layer * DD + n];
#pragma unroll
    for (int r = 0; r < 8; ++r) {
      size_t row = (size_t)(row0 + khi * 8 + r);
      a_out[row * DD + n] = acc1[r] + bias1;
      agg_out[row * DD + n] = acc3[r] + bias3 - wd[r] * acc2[r];
    }
  }
}

// ------- edge scatter: agg[dst] += edge_attr * a[src]  (32 lanes/edge) ------
__global__ __launch_bounds__(256) void edge_kernel(
    const int* __restrict__ ei, const float* __restrict__ ea,
    const float* __restrict__ a_buf, float* __restrict__ agg, int E) {
  int idx = blockIdx.x * blockDim.x + threadIdx.x;
  int e = idx >> 5;
  int lane = idx & 31;
  if (e >= E) return;
  int src = ei[e];
  int dst = ei[E + e];
  float w = ea[e];
  const float2 v = *(const float2*)(a_buf + (size_t)src * DD + lane * 2);
  float* p = agg + (size_t)dst * DD + lane * 2;
  atomAddF32(p, w * v.x);
  atomAddF32(p + 1, w * v.y);
}

// -------------------- h = relu(agg); refresh f16 copy -----------------------
__global__ __launch_bounds__(256) void update_kernel(
    const float* __restrict__ agg, float* __restrict__ h,
    _Float16* __restrict__ h16, int nd) {
  int i = blockIdx.x * blockDim.x + threadIdx.x;
  if (i >= nd) return;
  float v = agg[i];
  v = v > 0.f ? v : 0.f;
  h[i] = v;
  h16[i] = (_Float16)v;
}

// -------------------- per-graph mean pooling (atomics) ----------------------
__global__ __launch_bounds__(256) void pool_kernel(
    const float* __restrict__ h, const int* __restrict__ batch,
    float* __restrict__ gsum, float* __restrict__ gcnt, int N) {
  int i = blockIdx.x * blockDim.x + threadIdx.x;
  if (i >= N * DD) return;
  int n = i >> 6, d = i & 63;
  int g = batch[n];
  atomAddF32(&gsum[g * DD + d], h[i]);
  if (d == 0) atomAddF32(&gcnt[g], 1.0f);
}

// -------------------- MLP head: relu(gx@Wl1+bl1)@Wl2+bl2 --------------------
__global__ __launch_bounds__(32) void head_kernel(
    const float* __restrict__ gsum, const float* __restrict__ gcnt,
    const float* __restrict__ Wl1, const float* __restrict__ bl1,
    const float* __restrict__ Wl2, const float* __restrict__ bl2,
    float* __restrict__ pred) {
  int g = blockIdx.x;
  int j = threadIdx.x;  // 0..31
  __shared__ float gx[64];
  __shared__ float hid[32];
  float c = gcnt[g];
  c = c > 1.f ? c : 1.f;
  gx[j] = gsum[g * DD + j] / c;
  gx[j + 32] = gsum[g * DD + j + 32] / c;
  __syncthreads();
  float s = bl1[j];
#pragma unroll 8
  for (int d = 0; d < 64; ++d) s += gx[d] * Wl1[d * 32 + j];
  hid[j] = s > 0.f ? s : 0.f;
  __syncthreads();
  if (j < 3) {
    float o = bl2[j];
#pragma unroll 8
    for (int t = 0; t < 32; ++t) o += hid[t] * Wl2[t * 3 + j];
    pred[g * 3 + j] = o;
  }
}

extern "C" void kernel_launch(void* const* d_in, const int* in_sizes, int n_in,
                              void* d_out, int out_size, void* d_ws,
                              size_t ws_size, hipStream_t stream) {
  const float* x    = (const float*)d_in[0];
  const int*   ei   = (const int*)d_in[1];
  const float* ea   = (const float*)d_in[2];
  const int*   bat  = (const int*)d_in[3];
  const float* Wemb = (const float*)d_in[4];
  const float* bemb = (const float*)d_in[5];
  const float* W1   = (const float*)d_in[6];
  const float* b1   = (const float*)d_in[7];
  const float* W2   = (const float*)d_in[8];
  const float* W3   = (const float*)d_in[9];
  const float* b3   = (const float*)d_in[10];
  const float* Wl1  = (const float*)d_in[11];
  const float* bl1  = (const float*)d_in[12];
  const float* Wl2  = (const float*)d_in[13];
  const float* bl2  = (const float*)d_in[14];
  float* pred = (float*)d_out;

  const int N = NN, E = EE, G = GG;

  char* w = (char*)d_ws;
  auto wsAlloc = [&](size_t bytes) {
    char* p = w;
    w += (bytes + 255) & ~(size_t)255;
    return (void*)p;
  };
  float*    h     = (float*)wsAlloc((size_t)N * DD * 4);
  float*    abuf  = (float*)wsAlloc((size_t)N * DD * 4);
  float*    agg   = (float*)wsAlloc((size_t)N * DD * 4);
  _Float16* h16   = (_Float16*)wsAlloc((size_t)N * DD * 2);
  _Float16* wpack = (_Float16*)wsAlloc((size_t)3 * 12288 * 2);
  float*    wdeg  = (float*)wsAlloc((size_t)N * 4);
  float*    gsum  = (float*)wsAlloc((size_t)G * DD * 4);
  float*    gcnt  = (float*)wsAlloc((size_t)G * 4);

  hipMemsetAsync(wdeg, 0, (size_t)N * 4, stream);
  hipMemsetAsync(gsum, 0, (size_t)G * DD * 4, stream);
  hipMemsetAsync(gcnt, 0, (size_t)G * 4, stream);

  embed_kernel<<<(N * DD + 255) / 256, 256, 0, stream>>>(x, Wemb, bemb, h, h16, N);
  wdeg_kernel<<<(E + 255) / 256, 256, 0, stream>>>(ei, ea, wdeg, E);
  pack_weights_kernel<<<(2304 + 255) / 256, 256, 0, stream>>>(W1, W2, W3, wpack);

  const int ntiles = N / 16;  // 6250
  for (int l = 0; l < 3; ++l) {
    gemm3_kernel<<<(ntiles * 32 + 255) / 256, 256, 0, stream>>>(
        h16, wpack, b1, b3, wdeg, abuf, agg, l, ntiles);
    edge_kernel<<<(E * 32 + 255) / 256, 256, 0, stream>>>(ei, ea, abuf, agg, E);
    update_kernel<<<(N * DD + 255) / 256, 256, 0, stream>>>(agg, h, h16, N * DD);
  }

  pool_kernel<<<(N * DD + 255) / 256, 256, 0, stream>>>(h, bat, gsum, gcnt, N);
  head_kernel<<<G, 32, 0, stream>>>(gsum, gcnt, Wl1, bl1, Wl2, bl2, pred);
}
